// GATNet_augmented_24386824306813
// MI455X (gfx1250) — matrix operand order repaired and checked by reading, not compiled, not run on hardware
//
#include <hip/hip_runtime.h>
#include <hip/hip_bf16.h>

// ---------------------------------------------------------------------------
// CDNA5 (gfx1250) implementation of the GATNet+Conv reference.
// All dense GEMMs use v_wmma_f32_16x16x32_bf16 (wave32 WMMA), bf16 staged
// through LDS in fragment-friendly layout (ds_load_b128 per half-fragment).
// ---------------------------------------------------------------------------

typedef __attribute__((ext_vector_type(16))) __bf16 v16bf;
typedef __attribute__((ext_vector_type(8)))  float  v8f;

#define NN     50000
#define EE     200000
#define BB     512
#define F0_    78
#define H1_    10
#define C2_    128
#define EMB_   128
#define SEQ_   1000
#define NF_    32
#define KW_    8
#define LOUT_  121   // EMB - KW + 1

// ---------------- workspace layout (float units) ---------------------------
#define OFF_H1     ((size_t)0)
#define OFF_WST    ((size_t)0)
#define OFF_PST    ((size_t)256000)
#define OFF_XTIN   ((size_t)17033216)
#define OFF_XT     ((size_t)19015680)
#define OFF_XC     ((size_t)19081216)
#define OFF_F1     ((size_t)19212288)
#define OFF_F2     ((size_t)19736576)
#define OFF_OUT1   ((size_t)39000000)
#define OFF_ALS1   ((size_t)78000000)
#define OFF_ALD1   ((size_t)78500000)
#define OFF_AMAX1  ((size_t)79000000)
#define OFF_DEN1   ((size_t)79500000)
#define OFF_H2     ((size_t)80000000)
#define OFF_OUT2   ((size_t)86400000)
#define OFF_ALS2   ((size_t)92800000)
#define OFF_ALD2   ((size_t)92850000)
#define OFF_AMAX2  ((size_t)92900000)
#define OFF_DEN2   ((size_t)92950000)
#define OFF_G      ((size_t)93000000)
#define OFF_GFC    ((size_t)93065536)

// ---------------- helpers ---------------------------------------------------
__device__ __forceinline__ unsigned enc_f(float f) {
    unsigned u = __float_as_uint(f);
    return (f >= 0.f) ? (u | 0x80000000u) : ~u;
}
__device__ __forceinline__ float dec_f(unsigned u) {
    return (u & 0x80000000u) ? __uint_as_float(u & 0x7FFFFFFFu)
                             : __uint_as_float(~u);
}
__device__ __forceinline__ float leaky02(float x) { return x > 0.f ? x : 0.2f * x; }

__global__ void k_fill0(float* p, long n) {
    long i = (long)blockIdx.x * blockDim.x + threadIdx.x;
    long stride = (long)gridDim.x * blockDim.x;
    for (; i < n; i += stride) p[i] = 0.f;
}

// ---------------- WMMA GEMM core --------------------------------------------
// Block tile 64x128, BK=32, 256 threads = 8 wave32 waves.
// Wave (wr = w&3, wc = w>>2) owns rows 16*wr..+16, cols 64*wc..+64 (4 accs).
// LDS: sA[row][k] (64x32, stride 40 bf16 = 80B) and sBt[col][k] (128x32).
// Each lane's WMMA fragment = two 16B contiguous LDS chunks -> ds_load_b128.
#define TM 64
#define TN 128
#define TK 32
#define LDK 40   // padded k-stride in bf16 elems (80B, 16B-aligned)

union Frag16 { uint4 q[2]; v16bf v; };
union Pack8  { uint4 q;    __bf16 h[8]; };

template <int ACT, bool HAS_BIAS>   // ACT: 0 none, 1 relu, 2 elu
__global__ __launch_bounds__(256) void k_wmma_gemm(
    const float* __restrict__ A, const float* __restrict__ Bm,
    const float* __restrict__ bias, float* __restrict__ C,
    int M, int N, int K)
{
    __shared__ __align__(16) __bf16 sA[TM][LDK];
    __shared__ __align__(16) __bf16 sBt[TN][LDK];

    const int tid  = threadIdx.x;
    const int lane = tid & 31;
    const int wave = tid >> 5;
    const int wr = wave & 3;         // 0..3
    const int wc = wave >> 2;        // 0..1
    const int bm = blockIdx.y * TM;
    const int bn = blockIdx.x * TN;

    v8f acc[4] = {v8f{}, v8f{}, v8f{}, v8f{}};

    // per-thread staging coordinates (branch-free clamped loads)
    const int ar  = tid >> 2;        // A: row 0..63
    const int akq = tid & 3;         // A: k-quarter (8 k each)
    const int bnn = tid & 127;       // B: col-in-tile 0..127
    const int bkh = tid >> 7;        // B: k-half (16 k each)
    const int agr = bm + ar;
    const bool arok = (agr < M);
    const float* Arow = A + (size_t)(arok ? agr : 0) * K;
    const int bgc = bn + bnn;
    const bool bcok = (bgc < N);

    for (int k0 = 0; k0 < K; k0 += TK) {
        // ---- stage A tile: 8 bf16 per thread, one ds_store_b128 ----
        {
            Pack8 u;
            #pragma unroll
            for (int j = 0; j < 8; ++j) {
                int gk = k0 + akq * 8 + j;
                bool ok = arok && (gk < K);
                float v = Arow[ok ? gk : 0];
                u.h[j] = ok ? (__bf16)v : (__bf16)0.f;
            }
            *(uint4*)&sA[ar][akq * 8] = u.q;
        }
        // ---- stage B tile transposed: 16 bf16 per thread, 2x ds_store_b128 ----
        {
            Pack8 u0, u1;
            #pragma unroll
            for (int j = 0; j < 16; ++j) {
                int gk = k0 + bkh * 16 + j;
                bool ok = bcok && (gk < K);
                float v = Bm[ok ? ((size_t)gk * N + bgc) : 0];
                __bf16 h = ok ? (__bf16)v : (__bf16)0.f;
                if (j < 8) u0.h[j] = h; else u1.h[j - 8] = h;
            }
            *(uint4*)&sBt[bnn][bkh * 16]     = u0.q;
            *(uint4*)&sBt[bnn][bkh * 16 + 8] = u1.q;
        }
        __syncthreads();
        if (k0 + TK < K) {  // prefetch next A tile (global_prefetch_b8)
            if (arok) __builtin_prefetch(&Arow[k0 + TK + akq * 8], 0, 0);
        }

        // ---- A fragment: lanes 0-15 get k {kb..kb+7, 16+kb..}, kb=8*(lane>>4)
        Frag16 fa;
        {
            int arow = wr * 16 + (lane & 15);
            int kb   = (lane >> 4) * 8;
            fa.q[0] = *(const uint4*)&sA[arow][kb];
            fa.q[1] = *(const uint4*)&sA[arow][kb + 16];
        }
        // ---- 4 B fragments + 4 WMMAs ----
        #pragma unroll
        for (int j = 0; j < 4; ++j) {
            Frag16 fb;
            int col = wc * 64 + j * 16 + (lane & 15);
            int kg  = (lane >> 4) * 8;
            fb.q[0] = *(const uint4*)&sBt[col][kg];
            fb.q[1] = *(const uint4*)&sBt[col][kg + 16];
            acc[j] = __builtin_amdgcn_wmma_f32_16x16x32_bf16(
                false, fa.v, false, fb.v, (short)0, acc[j], false, false);
        }
        __syncthreads();
    }

    // ---- store: C layout — VGPR r: M = r + 8*(lane>>4), N = lane&15 ----
    int rbase = bm + wr * 16 + (lane >> 4) * 8;
    #pragma unroll
    for (int j = 0; j < 4; ++j) {
        int col = bn + wc * 64 + j * 16 + (lane & 15);
        if (col >= N) continue;
        float bv = HAS_BIAS ? bias[col] : 0.f;
        #pragma unroll
        for (int r = 0; r < 8; ++r) {
            int grow = rbase + r;
            if (grow >= M) continue;
            float v = acc[j][r] + bv;
            if (ACT == 1) v = fmaxf(v, 0.f);
            if (ACT == 2) v = v > 0.f ? v : (__expf(v) - 1.f);
            C[(size_t)grow * N + col] = v;
        }
    }
}

// ---------------- batched conv GEMM: P[b] = Wstack[256x1000] @ E_b[1000x128]
// E_b gathered on the fly from emb via target (im2col-free).  M tile 64,
// N = 128 = full TN, K = 1000.
__global__ __launch_bounds__(256) void k_conv_gemm(
    const float* __restrict__ Wstack, const float* __restrict__ emb,
    const int* __restrict__ target, float* __restrict__ P)
{
    __shared__ __align__(16) __bf16 sA[TM][LDK];
    __shared__ __align__(16) __bf16 sBt[TN][LDK];

    const int tid  = threadIdx.x;
    const int lane = tid & 31;
    const int wave = tid >> 5;
    const int wr = wave & 3;
    const int wc = wave >> 2;
    const int bm = blockIdx.y * TM;      // 0,64,128,192
    const int b  = blockIdx.z;
    const int* trow = target + (size_t)b * SEQ_;

    v8f acc[4] = {v8f{}, v8f{}, v8f{}, v8f{}};

    const int ar  = tid >> 2;
    const int akq = tid & 3;
    const int bnn = tid & 127;
    const int bkh = tid >> 7;
    const float* Arow = Wstack + (size_t)(bm + ar) * SEQ_;   // row always < 256

    for (int k0 = 0; k0 < SEQ_; k0 += TK) {
        {
            Pack8 u;
            #pragma unroll
            for (int j = 0; j < 8; ++j) {
                int gk = k0 + akq * 8 + j;
                bool ok = (gk < SEQ_);
                float v = Arow[ok ? gk : 0];
                u.h[j] = ok ? (__bf16)v : (__bf16)0.f;
            }
            *(uint4*)&sA[ar][akq * 8] = u.q;
        }
        {
            Pack8 u0, u1;
            #pragma unroll
            for (int j = 0; j < 16; ++j) {
                int gk = k0 + bkh * 16 + j;
                bool ok = (gk < SEQ_);
                int  tv = trow[ok ? gk : 0];
                float v = emb[(size_t)tv * EMB_ + bnn];
                __bf16 h = ok ? (__bf16)v : (__bf16)0.f;
                if (j < 8) u0.h[j] = h; else u1.h[j - 8] = h;
            }
            *(uint4*)&sBt[bnn][bkh * 16]     = u0.q;
            *(uint4*)&sBt[bnn][bkh * 16 + 8] = u1.q;
        }
        __syncthreads();

        Frag16 fa;
        {
            int arow = wr * 16 + (lane & 15);
            int kb   = (lane >> 4) * 8;
            fa.q[0] = *(const uint4*)&sA[arow][kb];
            fa.q[1] = *(const uint4*)&sA[arow][kb + 16];
        }
        #pragma unroll
        for (int j = 0; j < 4; ++j) {
            Frag16 fb;
            int col = wc * 64 + j * 16 + (lane & 15);
            int kg  = (lane >> 4) * 8;
            fb.q[0] = *(const uint4*)&sBt[col][kg];
            fb.q[1] = *(const uint4*)&sBt[col][kg + 16];
            acc[j] = __builtin_amdgcn_wmma_f32_16x16x32_bf16(
                false, fa.v, false, fb.v, (short)0, acc[j], false, false);
        }
        __syncthreads();
    }

    float* Pb = P + (size_t)b * 256 * EMB_;
    int rbase = bm + wr * 16 + (lane >> 4) * 8;
    #pragma unroll
    for (int j = 0; j < 4; ++j) {
        int col = wc * 64 + j * 16 + (lane & 15);
        #pragma unroll
        for (int r = 0; r < 8; ++r)
            Pb[(size_t)(rbase + r) * EMB_ + col] = acc[j][r];
    }
}

// ---------------- GAT auxiliary kernels -------------------------------------
__global__ void k_logits(const float* __restrict__ h, const float* __restrict__ a_src,
                         const float* __restrict__ a_dst, float* als, float* ald,
                         int n, int heads, int c)
{
    int t = blockIdx.x * blockDim.x + threadIdx.x;
    if (t >= n * heads) return;
    int node = t / heads, hh = t % heads;
    const float* hp  = h + (size_t)node * heads * c + (size_t)hh * c;
    const float* as_ = a_src + (size_t)hh * c;
    const float* ad_ = a_dst + (size_t)hh * c;
    float s = 0.f, d = 0.f;
    for (int i = 0; i < c; ++i) { float v = hp[i]; s += v * as_[i]; d += v * ad_[i]; }
    als[t] = s; ald[t] = d;
}

__global__ void k_edge_max(const int* __restrict__ ei, int E, int n,
                           const float* __restrict__ als, const float* __restrict__ ald,
                           unsigned* __restrict__ amax, int heads)
{
    int e = blockIdx.x * blockDim.x + threadIdx.x;
    if (e >= E + n) return;
    int src = (e < E) ? ei[e]     : (e - E);
    int dst = (e < E) ? ei[E + e] : (e - E);
    for (int hh = 0; hh < heads; ++hh) {
        float al = leaky02(als[src * heads + hh] + ald[dst * heads + hh]);
        atomicMax(&amax[dst * heads + hh], enc_f(al));
    }
}

__global__ void k_edge_den(const int* __restrict__ ei, int E, int n,
                           const float* __restrict__ als, const float* __restrict__ ald,
                           const unsigned* __restrict__ amax, float* __restrict__ den,
                           int heads)
{
    int e = blockIdx.x * blockDim.x + threadIdx.x;
    if (e >= E + n) return;
    int src = (e < E) ? ei[e]     : (e - E);
    int dst = (e < E) ? ei[E + e] : (e - E);
    for (int hh = 0; hh < heads; ++hh) {
        float al = leaky02(als[src * heads + hh] + ald[dst * heads + hh]);
        float ex = __expf(al - dec_f(amax[dst * heads + hh]));
        atomicAdd(&den[dst * heads + hh], ex);
    }
}

__global__ void k_edge_scatter(const int* __restrict__ ei, int E, int n,
                               const float* __restrict__ als, const float* __restrict__ ald,
                               const unsigned* __restrict__ amax, const float* __restrict__ den,
                               const float* __restrict__ h, float* __restrict__ out,
                               int heads, int c)
{
    int t = blockIdx.x * blockDim.x + threadIdx.x;
    if (t >= (E + n) * heads) return;
    int e  = t / heads, hh = t % heads;
    int src = (e < E) ? ei[e]     : (e - E);
    int dst = (e < E) ? ei[E + e] : (e - E);
    float al = leaky02(als[src * heads + hh] + ald[dst * heads + hh]);
    float ex = __expf(al - dec_f(amax[dst * heads + hh]));
    float alpha = ex / (den[dst * heads + hh] + 1e-16f);
    const float* hs = h   + (size_t)src * heads * c + (size_t)hh * c;
    float*       od = out + (size_t)dst * heads * c + (size_t)hh * c;
    for (int i = 0; i < c; ++i) atomicAdd(&od[i], hs[i] * alpha);
}

template <int ACT>  // 1 relu, 2 elu
__global__ void k_bias_act(float* __restrict__ v, const float* __restrict__ bias,
                           int cols, long total)
{
    long i = (long)blockIdx.x * blockDim.x + threadIdx.x;
    if (i >= total) return;
    float x = v[i] + bias[i % cols];
    if (ACT == 1) x = fmaxf(x, 0.f);
    else          x = x > 0.f ? x : (__expf(x) - 1.f);
    v[i] = x;
}

__global__ void k_pool_max(const float* __restrict__ h, const int* __restrict__ batch,
                           float* __restrict__ g, int n)
{
    long t = (long)blockIdx.x * blockDim.x + threadIdx.x;
    if (t >= (long)n * C2_) return;
    int node = (int)(t / C2_), ch = (int)(t % C2_);
    float v = h[t];  // post-ReLU, >= 0 -> uint bits order == float order
    atomicMax((unsigned*)&g[(size_t)batch[node] * C2_ + ch], __float_as_uint(v));
}

// ---------------- conv stage helpers ----------------------------------------
__global__ void k_wstack(const float* __restrict__ conv_w, float* __restrict__ Wstack)
{
    int t = blockIdx.x * blockDim.x + threadIdx.x;       // 256*1000
    if (t >= 256 * SEQ_) return;
    int row = t / SEQ_, s = t % SEQ_;
    int k = row >> 5, f = row & 31;
    Wstack[(size_t)row * SEQ_ + s] = conv_w[(size_t)f * SEQ_ * KW_ + (size_t)s * KW_ + k];
}

__global__ void k_conv_epi(const float* __restrict__ P, const float* __restrict__ conv_b,
                           float* __restrict__ xtin)
{
    int t = blockIdx.x * blockDim.x + threadIdx.x;       // B*32*121
    if (t >= BB * NF_ * LOUT_) return;
    int l = t % LOUT_;
    int f = (t / LOUT_) % NF_;
    int b = t / (LOUT_ * NF_);
    const float* Pb = P + (size_t)b * 256 * EMB_;
    float s = conv_b[f];
    #pragma unroll
    for (int k = 0; k < KW_; ++k) s += Pb[(size_t)(k * 32 + f) * EMB_ + (l + k)];
    xtin[(size_t)b * (NF_ * LOUT_) + f * LOUT_ + l] = fmaxf(s, 0.f);
}

__global__ void k_concat(const float* __restrict__ gfc, const float* __restrict__ xt,
                         float* __restrict__ xc)
{
    int t = blockIdx.x * blockDim.x + threadIdx.x;       // B*256
    if (t >= BB * 256) return;
    int b = t >> 8, j = t & 255;
    xc[t] = (j < 128) ? gfc[b * 128 + j] : xt[b * 128 + (j - 128)];
}

__global__ void k_final(const float* __restrict__ f2, const float* __restrict__ out_w,
                        const float* __restrict__ out_b, float* __restrict__ out)
{
    int b = blockIdx.x * blockDim.x + threadIdx.x;
    if (b >= BB) return;
    float s = 0.f;
    for (int i = 0; i < 256; ++i) s += f2[b * 256 + i] * out_w[i];
    out[b] = s + out_b[0];
}

// ---------------------------------------------------------------------------
extern "C" void kernel_launch(void* const* d_in, const int* in_sizes, int n_in,
                              void* d_out, int out_size, void* d_ws, size_t ws_size,
                              hipStream_t stream)
{
    const float* x       = (const float*)d_in[0];
    const int*   ei      = (const int*)  d_in[1];
    const int*   batch   = (const int*)  d_in[2];
    const int*   target  = (const int*)  d_in[3];
    const float* W1      = (const float*)d_in[4];
    const float* a_src1  = (const float*)d_in[5];
    const float* a_dst1  = (const float*)d_in[6];
    const float* b1      = (const float*)d_in[7];
    const float* W2      = (const float*)d_in[8];
    const float* a_src2  = (const float*)d_in[9];
    const float* a_dst2  = (const float*)d_in[10];
    const float* b2      = (const float*)d_in[11];
    const float* fc_g1_w = (const float*)d_in[12];
    const float* fc_g1_b = (const float*)d_in[13];
    const float* emb     = (const float*)d_in[14];
    const float* conv_w  = (const float*)d_in[15];
    const float* conv_b  = (const float*)d_in[16];
    const float* fc_xt_w = (const float*)d_in[17];
    const float* fc_xt_b = (const float*)d_in[18];
    const float* fc1_w   = (const float*)d_in[19];
    const float* fc1_b   = (const float*)d_in[20];
    const float* fc2_w   = (const float*)d_in[21];
    const float* fc2_b   = (const float*)d_in[22];
    const float* out_w   = (const float*)d_in[23];
    const float* out_b   = (const float*)d_in[24];
    float* out = (float*)d_out;

    float* ws = (float*)d_ws;
    float*    h1    = ws + OFF_H1;
    float*    out1  = ws + OFF_OUT1;
    float*    als1  = ws + OFF_ALS1;
    float*    ald1  = ws + OFF_ALD1;
    unsigned* amax1 = (unsigned*)(ws + OFF_AMAX1);
    float*    den1  = ws + OFF_DEN1;
    float*    h2    = ws + OFF_H2;
    float*    out2  = ws + OFF_OUT2;
    float*    als2  = ws + OFF_ALS2;
    float*    ald2  = ws + OFF_ALD2;
    unsigned* amax2 = (unsigned*)(ws + OFF_AMAX2);
    float*    den2  = ws + OFF_DEN2;
    float*    g     = ws + OFF_G;
    float*    gfc   = ws + OFF_GFC;
    float*    Wst   = ws + OFF_WST;
    float*    Pst   = ws + OFF_PST;
    float*    xtin  = ws + OFF_XTIN;
    float*    xt    = ws + OFF_XT;
    float*    xc    = ws + OFF_XC;
    float*    f1    = ws + OFF_F1;
    float*    f2    = ws + OFF_F2;

    const int N = NN, E = EE, EP = EE + NN;
    auto cdiv = [](long a, long b) { return (int)((a + b - 1) / b); };

    // ---- zero-init accumulators ----
    k_fill0<<<2048, 256, 0, stream>>>(out1, (long)N * 780);
    k_fill0<<<512,  256, 0, stream>>>((float*)amax1, (long)N * H1_);
    k_fill0<<<512,  256, 0, stream>>>(den1, (long)N * H1_);
    k_fill0<<<1024, 256, 0, stream>>>(out2, (long)N * C2_);
    k_fill0<<<64,   256, 0, stream>>>((float*)amax2, (long)N);
    k_fill0<<<64,   256, 0, stream>>>(den2, (long)N);
    k_fill0<<<64,   256, 0, stream>>>(g, (long)BB * C2_);

    // ---- GAT layer 1 ----
    {
        dim3 gg(cdiv(780, TN), cdiv(N, TM));
        k_wmma_gemm<0, false><<<gg, 256, 0, stream>>>(x, W1, nullptr, h1, N, 780, F0_);
    }
    k_logits<<<cdiv((long)N * H1_, 256), 256, 0, stream>>>(h1, a_src1, a_dst1, als1, ald1, N, H1_, F0_);
    k_edge_max    <<<cdiv(EP, 256), 256, 0, stream>>>(ei, E, N, als1, ald1, amax1, H1_);
    k_edge_den    <<<cdiv(EP, 256), 256, 0, stream>>>(ei, E, N, als1, ald1, amax1, den1, H1_);
    k_edge_scatter<<<cdiv((long)EP * H1_, 256), 256, 0, stream>>>(ei, E, N, als1, ald1, amax1, den1,
                                                                  h1, out1, H1_, F0_);
    k_bias_act<2><<<cdiv((long)N * 780, 256), 256, 0, stream>>>(out1, b1, 780, (long)N * 780);

    // ---- GAT layer 2 ----
    {
        dim3 gg(cdiv(C2_, TN), cdiv(N, TM));
        k_wmma_gemm<0, false><<<gg, 256, 0, stream>>>(out1, W2, nullptr, h2, N, C2_, 780);
    }
    k_logits<<<cdiv(N, 256), 256, 0, stream>>>(h2, a_src2, a_dst2, als2, ald2, N, 1, C2_);
    k_edge_max    <<<cdiv(EP, 256), 256, 0, stream>>>(ei, E, N, als2, ald2, amax2, 1);
    k_edge_den    <<<cdiv(EP, 256), 256, 0, stream>>>(ei, E, N, als2, ald2, amax2, den2, 1);
    k_edge_scatter<<<cdiv(EP, 256), 256, 0, stream>>>(ei, E, N, als2, ald2, amax2, den2,
                                                      h2, out2, 1, C2_);
    k_bias_act<1><<<cdiv((long)N * C2_, 256), 256, 0, stream>>>(out2, b2, C2_, (long)N * C2_);

    // ---- global max pool + graph fc ----
    k_pool_max<<<cdiv((long)N * C2_, 256), 256, 0, stream>>>(out2, batch, g, N);
    {
        dim3 gg(1, cdiv(BB, TM));
        k_wmma_gemm<1, true><<<gg, 256, 0, stream>>>(g, fc_g1_w, fc_g1_b, gfc, BB, C2_, C2_);
    }

    // ---- conv stage (reuses h1 region; runs after GAT1 is done with h1) ----
    k_wstack<<<cdiv(256 * SEQ_, 256), 256, 0, stream>>>(conv_w, Wst);
    {
        dim3 gg(1, 4, BB);  // N=128 (1 tile), M=256 (4 tiles), per batch
        k_conv_gemm<<<gg, 256, 0, stream>>>(Wst, emb, target, Pst);
    }
    k_conv_epi<<<cdiv((long)BB * NF_ * LOUT_, 256), 256, 0, stream>>>(Pst, conv_b, xtin);
    {
        dim3 gg(1, cdiv(BB, TM));
        k_wmma_gemm<0, true><<<gg, 256, 0, stream>>>(xtin, fc_xt_w, fc_xt_b, xt, BB, C2_, NF_ * LOUT_);
    }

    // ---- MLP head ----
    k_concat<<<cdiv(BB * 256, 256), 256, 0, stream>>>(gfc, xt, xc);
    {
        dim3 gg(cdiv(1024, TN), cdiv(BB, TM));
        k_wmma_gemm<1, true><<<gg, 256, 0, stream>>>(xc, fc1_w, fc1_b, f1, BB, 1024, 256);
    }
    {
        dim3 gg(cdiv(256, TN), cdiv(BB, TM));
        k_wmma_gemm<1, true><<<gg, 256, 0, stream>>>(f1, fc2_w, fc2_b, f2, BB, 256, 1024);
    }
    k_final<<<cdiv(BB, 256), 256, 0, stream>>>(f2, out_w, out_b, out);
}
